// WL4PatternNet_18073222382051
// MI455X (gfx1250) — matrix operand order repaired and checked by reading, not compiled, not run on hardware
//
#include <hip/hip_runtime.h>

// ---------------- problem constants (match reference) ----------------
#define NN     200000          // nodes
#define DEGQ   16              // uniform degree
#define SS     128             // subgraphs
#define SUBN   512             // nodes per subgraph
#define LLAY   3               // WL iterations
#define HTBITS 20
#define HTSIZE (1u << HTBITS)  // hash table entries (load factor ~0.19)
#define HTMASK (HTSIZE - 1u)

#define SCAN_TILE 1024         // elements per scan block (256 thr x 4)

typedef __attribute__((ext_vector_type(16))) _Float16 v16h;
typedef __attribute__((ext_vector_type(8)))  float    v8f;

// ---------------- helpers ----------------
__device__ __forceinline__ unsigned long long mix64(unsigned long long h) {
  h ^= h >> 33; h *= 0xff51afd7ed558ccdULL;
  h ^= h >> 33; h *= 0xc4ceb9fe1a85ec53ULL;
  h ^= h >> 33; return h;
}

// Batcher odd-even mergesort network for 16 elements (63 comparators),
// constant indices only -> stays entirely in registers.
#define CE(i, j) { int _a = t[i], _b = t[j]; t[i] = _a < _b ? _a : _b; t[j] = _a < _b ? _b : _a; }
__device__ __forceinline__ void sort16(int* t) {
  CE(0,1) CE(2,3) CE(4,5) CE(6,7) CE(8,9) CE(10,11) CE(12,13) CE(14,15)
  CE(0,2) CE(1,3) CE(4,6) CE(5,7) CE(8,10) CE(9,11) CE(12,14) CE(13,15)
  CE(1,2) CE(5,6) CE(9,10) CE(13,14)
  CE(0,4) CE(1,5) CE(2,6) CE(3,7) CE(8,12) CE(9,13) CE(10,14) CE(11,15)
  CE(2,4) CE(3,5) CE(10,12) CE(11,13)
  CE(1,2) CE(3,4) CE(5,6) CE(9,10) CE(11,12) CE(13,14)
  CE(0,8) CE(1,9) CE(2,10) CE(3,11) CE(4,12) CE(5,13) CE(6,14) CE(7,15)
  CE(4,8) CE(5,9) CE(6,10) CE(7,11)
  CE(2,4) CE(3,5) CE(6,8) CE(7,9) CE(10,12) CE(11,13)
  CE(1,2) CE(3,4) CE(5,6) CE(7,8) CE(9,10) CE(11,12) CE(13,14)
}
#undef CE

// ---------------- kernel 1: initial colors via WMMA ----------------
// argmax of one-hot row == dot(row, [0..31]).  One 16x16x32 f16 WMMA handles
// 16 nodes: A = x tile (16 nodes x 32 classes), B[k][n] = k for all n, so
// D[m][n] = color[m] for every n (result robust in N).  Exact in f16.
__global__ void wl_init_colors(const float* __restrict__ x, int* __restrict__ color) {
  int wave = (int)((blockIdx.x * blockDim.x + threadIdx.x) >> 5);
  int lane = (int)(threadIdx.x & 31);
  int base = wave * 16;                 // 16 nodes per wave
  if (base >= NN) return;               // uniform per wave (EXEC all-1 for WMMA)

  int m = lane & 15;                    // row within tile
  const float* xr = x + (size_t)(base + m) * 32;

  // ISA 16-bit A 16x32 layout: lanes 0-15 hold K={0..7,16..23} of row m,
  // lanes 16-31 hold K={8..15,24..31} of row m.
  int k0 = (lane < 16) ? 0 : 8;
  int k1 = (lane < 16) ? 16 : 24;
  v16h a;
#pragma unroll
  for (int j = 0; j < 8; ++j) a[j] = (_Float16)xr[k0 + j];
#pragma unroll
  for (int j = 0; j < 8; ++j) a[8 + j] = (_Float16)xr[k1 + j];

  // B 32x16 16-bit layout: lane gives N=lane%16; half-slot j gives
  // K = (lane>=16 ? 8 : 0) + j + (j>=8 ? 8 : 0) (same K order as A).  Value only depends on K -> B[k][n] = k.
  int kb = (lane < 16) ? 0 : 8;
  v16h b;
#pragma unroll
  for (int j = 0; j < 16; ++j) b[j] = (_Float16)(float)(kb + j + (j >= 8 ? 8 : 0));

  v8f c = {};
  c = __builtin_amdgcn_wmma_f32_16x16x32_f16(false, a, false, b, (short)0, c, false, false);

  // D layout: VGPR r, lanes 0-15 -> (M=r, N=lane); lanes 16-31 -> (M=r+8, N=lane-16).
  // Read column N=0 from lanes 0 and 16.
  if (m == 0) {
    int mb = (lane < 16) ? 0 : 8;
#pragma unroll
    for (int r = 0; r < 8; ++r)
      color[base + mb + r] = (int)(c[r] + 0.5f);
  }
}

// ---------------- kernel 2: WL key build + hash insert ----------------
__global__ void wl_insert(const int* __restrict__ color, const int* __restrict__ nbr,
                          unsigned long long* __restrict__ keys, unsigned* __restrict__ minv,
                          unsigned* __restrict__ slotOf) {
  int i = blockIdx.x * blockDim.x + threadIdx.x;
  if (i >= NN) return;

  const int4* nb4 = (const int4*)(nbr + (size_t)i * DEGQ);
  int4 n0 = nb4[0], n1 = nb4[1], n2 = nb4[2], n3 = nb4[3];

  int t[16];
  t[0]  = color[n0.x]; t[1]  = color[n0.y]; t[2]  = color[n0.z]; t[3]  = color[n0.w];
  t[4]  = color[n1.x]; t[5]  = color[n1.y]; t[6]  = color[n1.z]; t[7]  = color[n1.w];
  t[8]  = color[n2.x]; t[9]  = color[n2.y]; t[10] = color[n2.z]; t[11] = color[n2.w];
  t[12] = color[n3.x]; t[13] = color[n3.y]; t[14] = color[n3.z]; t[15] = color[n3.w];
  sort16(t);

  unsigned long long h = mix64(0x9E3779B97F4A7C15ULL ^ (unsigned)color[i]);
#pragma unroll
  for (int j = 0; j < 16; ++j) h = mix64(h ^ (unsigned long long)(unsigned)t[j]);
  if (h == 0ULL) h = 1ULL;

  unsigned slot = (unsigned)(h >> 20) & HTMASK;
  for (;;) {
    unsigned long long prev = atomicCAS(&keys[slot], 0ULL, h);
    if (prev == 0ULL || prev == h) {      // claimed or joined: unique slot per key
      atomicMin(&minv[slot], (unsigned)i);
      slotOf[i] = slot;
      break;
    }
    slot = (slot + 1u) & HTMASK;
  }
}

// ---------------- kernel 3: representative flags ----------------
__global__ void wl_flag(const unsigned* __restrict__ slotOf, const unsigned* __restrict__ minv,
                        unsigned* __restrict__ flags) {
  int i = blockIdx.x * blockDim.x + threadIdx.x;
  if (i >= NN) return;
  flags[i] = (minv[slotOf[i]] == (unsigned)i) ? 1u : 0u;
}

// ---------------- scan (3-phase exclusive prefix sum) ----------------
__global__ void scan_blocks(const unsigned* __restrict__ flags, unsigned* __restrict__ exScan,
                            unsigned* __restrict__ blockSums) {
  __shared__ unsigned s[256];
  int t = threadIdx.x;
  int base = blockIdx.x * SCAN_TILE + t * 4;
  unsigned v[4]; unsigned sum = 0;
#pragma unroll
  for (int j = 0; j < 4; ++j) { int idx = base + j; v[j] = (idx < NN) ? flags[idx] : 0u; sum += v[j]; }
  s[t] = sum; __syncthreads();
  for (int off = 1; off < 256; off <<= 1) {
    unsigned add = (t >= off) ? s[t - off] : 0u;
    __syncthreads(); s[t] += add; __syncthreads();
  }
  unsigned excl = (t == 0) ? 0u : s[t - 1];
  if (t == 255) blockSums[blockIdx.x] = s[255];
  unsigned run = excl;
#pragma unroll
  for (int j = 0; j < 4; ++j) { int idx = base + j; if (idx < NN) exScan[idx] = run; run += v[j]; }
}

__global__ void scan_sums(unsigned* __restrict__ blockSums, int nb) {
  __shared__ unsigned s[256];
  int t = threadIdx.x;
  unsigned v = (t < nb) ? blockSums[t] : 0u;
  s[t] = v; __syncthreads();
  for (int off = 1; off < 256; off <<= 1) {
    unsigned add = (t >= off) ? s[t - off] : 0u;
    __syncthreads(); s[t] += add; __syncthreads();
  }
  unsigned excl = (t == 0) ? 0u : s[t - 1];
  if (t < nb) blockSums[t] = excl;
}

// ---------------- kernel 4: relabel to insertion-order ids ----------------
__global__ void wl_relabel(const unsigned* __restrict__ slotOf, const unsigned* __restrict__ minv,
                           const unsigned* __restrict__ exScan, const unsigned* __restrict__ blockSums,
                           int* __restrict__ colorNew, float* __restrict__ outLayer) {
  int i = blockIdx.x * blockDim.x + threadIdx.x;
  if (i >= NN) return;
  unsigned rep = minv[slotOf[i]];                      // group's first-occurring node
  unsigned c = exScan[rep] + blockSums[rep >> 10];     // rank among representatives
  colorNew[i] = (int)c;
  outLayer[i] = (float)c;
}

// ---------------- kernel 5: per-subgraph LDS histogram + L2 norm ----------------
__global__ void sub_hist(const int* __restrict__ color, const int* __restrict__ subNodes,
                         float* __restrict__ hist) {
  __shared__ int hkey[1024];
  __shared__ int hcnt[1024];
  __shared__ unsigned sumsq;
  int s = blockIdx.x;
  int t = threadIdx.x;                                 // blockDim == 512
  for (int e = t; e < 1024; e += 512) { hkey[e] = -1; hcnt[e] = 0; }
  if (t == 0) sumsq = 0u;
  __syncthreads();

  int c = color[subNodes[s * SUBN + t]];
  unsigned slot = (((unsigned)c * 2654435761u) >> 12) & 1023u;
  for (;;) {
    int prev = atomicCAS(&hkey[slot], -1, c);
    if (prev == -1 || prev == c) { atomicAdd(&hcnt[slot], 1); break; }
    slot = (slot + 1u) & 1023u;
  }
  __syncthreads();

  unsigned local = 0;
  for (int e = t; e < 1024; e += 512) { unsigned cc = (unsigned)hcnt[e]; local += cc * cc; }
  atomicAdd(&sumsq, local);                            // integer -> deterministic
  __syncthreads();

  float scale = 1.0f / sqrtf((float)sumsq);
  float* row = hist + (size_t)s * NN;
  for (int e = t; e < 1024; e += 512) {
    int k = hkey[e];
    if (k >= 0) row[k] = (float)hcnt[e] * scale;       // sparse overlay on zeroed row
  }
}

// ---------------- launch ----------------
extern "C" void kernel_launch(void* const* d_in, const int* in_sizes, int n_in,
                              void* d_out, int out_size, void* d_ws, size_t ws_size,
                              hipStream_t stream) {
  const float* x    = (const float*)d_in[0];           // [N, 32] one-hot
  const int*   edge = (const int*)d_in[1];             // [2, N*DEG]; row0 = sources = nbr lists
  const int*   sub  = (const int*)d_in[2];             // [S, SUB]
  float* outColors  = (float*)d_out;                   // [3, N]
  float* outHist    = (float*)d_out + (size_t)LLAY * NN;   // [S, N]

  // edge_index[1] = repeat(arange(N), DEG) is already sorted and argsort is
  // stable, so nbr_idx == edge_index[0].reshape(N, DEG) directly.
  const int* nbr = edge;

  // workspace carve-out (256B aligned), ~16.6 MB total
  char* w = (char*)d_ws; size_t off = 0;
  auto carve = [&](size_t bytes) { void* p = w + off; off = (off + bytes + 255) & ~(size_t)255; return p; };
  int*                colorA    = (int*)carve((size_t)NN * 4);
  int*                colorB    = (int*)carve((size_t)NN * 4);
  unsigned long long* keys      = (unsigned long long*)carve((size_t)HTSIZE * 8);
  unsigned*           minv      = (unsigned*)carve((size_t)HTSIZE * 4);
  unsigned*           slotOf    = (unsigned*)carve((size_t)NN * 4);
  unsigned*           flags     = (unsigned*)carve((size_t)NN * 4);
  unsigned*           exScan    = (unsigned*)carve((size_t)NN * 4);
  unsigned*           blockSums = (unsigned*)carve(256 * 4);
  (void)in_sizes; (void)n_in; (void)out_size; (void)ws_size;

  const int NT = 256;
  const int NB = (NN + NT - 1) / NT;                   // 782
  const int SCAN_NB = (NN + SCAN_TILE - 1) / SCAN_TILE; // 196

  // initial colors: one WMMA per 16 nodes -> 12500 waves
  {
    int waves = NN / 16;
    int blocks = (waves * 32 + NT - 1) / NT;           // 1563
    wl_init_colors<<<blocks, NT, 0, stream>>>(x, colorA);
  }

  int* cur = colorA; int* nxt = colorB;
  for (int l = 0; l < LLAY; ++l) {
    hipMemsetAsync(keys, 0x00, (size_t)HTSIZE * 8, stream);
    hipMemsetAsync(minv, 0xFF, (size_t)HTSIZE * 4, stream);
    wl_insert<<<NB, NT, 0, stream>>>(cur, nbr, keys, minv, slotOf);
    wl_flag<<<NB, NT, 0, stream>>>(slotOf, minv, flags);
    scan_blocks<<<SCAN_NB, 256, 0, stream>>>(flags, exScan, blockSums);
    scan_sums<<<1, 256, 0, stream>>>(blockSums, SCAN_NB);
    wl_relabel<<<NB, NT, 0, stream>>>(slotOf, minv, exScan, blockSums,
                                      nxt, outColors + (size_t)l * NN);
    int* tmp = cur; cur = nxt; nxt = tmp;
  }

  // histogram of last layer: zero 102.4 MB once, then sparse normalized overlay
  hipMemsetAsync(outHist, 0, (size_t)SS * NN * 4, stream);
  sub_hist<<<SS, SUBN, 0, stream>>>(cur, sub, outHist);
}
